// MultiHeadAttention_6536940225113
// MI455X (gfx1250) — compile-verified, matrix operand order checked by reading, not run on hardware
//
#include <hip/hip_runtime.h>

// ---------------------------------------------------------------------------
// MI455X (gfx1250) multi-head attention, bf16 WMMA pipeline.
//   D=1024, H=16, Dh=64, B=2, S=2048.
// ---------------------------------------------------------------------------

#define D_MODEL 1024
#define NUM_HEADS 16
#define HEAD_DIM 64
#define BATCH 2
#define SEQ 2048
#define MROWS (BATCH * SEQ)          // 4096
#define INV_SCALE 0.125f             // 1/sqrt(64)

typedef __attribute__((ext_vector_type(16))) __bf16 v16bf;
typedef __attribute__((ext_vector_type(8)))  float  v8f;
typedef unsigned short ushort_t;
typedef unsigned int   uint_t;

union Frag16 {
    v16bf    bf;
    uint4    u[2];
    ushort_t us[16];
};

__device__ __forceinline__ ushort_t f32_to_bf16_rne(float x) {
    uint_t u = __builtin_bit_cast(uint_t, x);
    u += 0x7FFFu + ((u >> 16) & 1u);          // round-to-nearest-even
    return (ushort_t)(u >> 16);
}

__device__ __forceinline__ v8f wmma_bf16(v16bf a, v16bf b, v8f c) {
    return __builtin_amdgcn_wmma_f32_16x16x32_bf16(
        /*neg_a=*/false, a, /*neg_b=*/false, b,
        /*c_mod=*/(short)0, c, /*reuse_a=*/false, /*reuse_b=*/false);
}

// Load a 16x32 bf16 WMMA fragment from K-contiguous row-major storage.
// Lane l: row = l&15 ; holds K = [(l>>4)*8 .. +7] and [16+(l>>4)*8 .. +7]
// (matches CDNA5 ISA 16-bit A-matrix layout; B^T columns load identically).
__device__ __forceinline__ v16bf load_frag_kmajor(const ushort_t* __restrict__ base,
                                                  int ld, int row0, int k0, int lane) {
    const ushort_t* p = base + (size_t)(row0 + (lane & 15)) * ld
                             + k0 + ((lane >> 4) << 3);
    Frag16 f;
    f.u[0] = *(const uint4*)(p);
    f.u[1] = *(const uint4*)(p + 16);
    return f.bf;
}

// ---------------------------------------------------------------------------
// 1) f32 -> bf16 conversion
// ---------------------------------------------------------------------------
__global__ void mha_cvt_bf16(const float* __restrict__ src,
                             ushort_t* __restrict__ dst, int n) {
    for (int i = blockIdx.x * blockDim.x + threadIdx.x; i < n;
         i += gridDim.x * blockDim.x)
        dst[i] = f32_to_bf16_rne(src[i]);
}

// ---------------------------------------------------------------------------
// 2) GEMM: Y = X @ W^T  (WMMA bf16, f32 accum).
//    X [4096,1024] bf16 row-major, W [1024,1024] bf16 row-major (N x K).
//    Block: 128 threads (4 waves); tile 16(M) x 64(N); wave w -> N-subtile w.
//    MODE 0: bf16 out, head layout      [B,H,S,Dh]
//    MODE 1: bf16 out, head-T layout    [B,H,Dh,S]   (for V -> K-major P@V)
//    MODE 2: f32 out + bias, row-major  [M,N]
// ---------------------------------------------------------------------------
template <int MODE>
__global__ __launch_bounds__(128)
void mha_gemm(const ushort_t* __restrict__ X,
              const ushort_t* __restrict__ W,
              const float* __restrict__ bias,
              void* __restrict__ outp) {
    const int lane = threadIdx.x & 31;
    const int wv   = threadIdx.x >> 5;
    const int m0   = blockIdx.x * 16;
    const int n0   = blockIdx.y * 64 + wv * 16;

    v8f acc = {};
    for (int k0 = 0; k0 < D_MODEL; k0 += 32) {
        if (k0 + 32 < D_MODEL) {   // -> global_prefetch_b8
            __builtin_prefetch(X + (size_t)(m0 + (lane & 15)) * D_MODEL + k0 + 32, 0, 0);
            __builtin_prefetch(W + (size_t)(n0 + (lane & 15)) * D_MODEL + k0 + 32, 0, 0);
        }
        v16bf a = load_frag_kmajor(X, D_MODEL, m0, k0, lane);
        v16bf b = load_frag_kmajor(W, D_MODEL, n0, k0, lane);
        acc = wmma_bf16(a, b, acc);
    }

    // C/D layout: lane l -> col n0 + (l&15); rows (l>>4)*8 + i, i=0..7
    const int c  = lane & 15;
    const int rb = (lane >> 4) * 8;
    const int n  = n0 + c;
    if (MODE == 2) {
        float* out = (float*)outp;
        const float bo = bias[n];
#pragma unroll
        for (int i = 0; i < 8; ++i) {
            const int m = m0 + rb + i;
            out[(size_t)m * D_MODEL + n] = acc[i] + bo;
        }
    } else {
        ushort_t* out = (ushort_t*)outp;
        const int h  = n >> 6;         // / HEAD_DIM
        const int dh = n & 63;
#pragma unroll
        for (int i = 0; i < 8; ++i) {
            const int m  = m0 + rb + i;
            const int bi = m >> 11;    // / SEQ
            const int s  = m & 2047;
            const size_t headOff = ((size_t)bi * NUM_HEADS + h);
            const size_t idx = (MODE == 0)
                ? (headOff * SEQ + s) * HEAD_DIM + dh
                : (headOff * HEAD_DIM + dh) * SEQ + s;
            out[idx] = f32_to_bf16_rne(acc[i]);
        }
    }
    (void)bias;
}

// ---------------------------------------------------------------------------
// 3) Fused attention: one block = 16 query rows of one (b,h).
//    Full 16x2048 score strip resident in LDS (128KB of the 320KB WGP LDS).
//    8 waves: QK^T WMMA -> exact softmax (wave32 shuffles, float4 streaming)
//    -> attn_weights b128 stores -> P@V WMMA (all 8 waves, split-K) -> bf16 ctx.
// ---------------------------------------------------------------------------
#define SPAD (SEQ + 8)
__global__ __launch_bounds__(256)
void mha_attn(const ushort_t* __restrict__ Qh,
              const ushort_t* __restrict__ Kh,
              const ushort_t* __restrict__ Vt,   // [B,H,Dh,S]
              float* __restrict__ attnw,
              ushort_t* __restrict__ ctx) {
    __shared__ float sc[16][SPAD];
    __shared__ float red[4][16][16];   // split-K partial context tiles

    const int lane = threadIdx.x & 31;
    const int wv   = threadIdx.x >> 5;
    const int q0   = blockIdx.x * 16;
    const int h    = blockIdx.y;
    const int b    = blockIdx.z;

    const size_t head = (size_t)b * NUM_HEADS + h;
    const ushort_t* Q = Qh + head * SEQ * HEAD_DIM;
    const ushort_t* K = Kh + head * SEQ * HEAD_DIM;
    const ushort_t* V = Vt + head * HEAD_DIM * SEQ;  // transposed: [Dh,S]

    // Q fragments for this 16-row strip (Dh=64 -> two K=32 chunks), kept live.
    const v16bf aq0 = load_frag_kmajor(Q, HEAD_DIM, q0, 0,  lane);
    const v16bf aq1 = load_frag_kmajor(Q, HEAD_DIM, q0, 32, lane);

    // ---- scores = (Q K^T) * INV_SCALE, into LDS ----
    for (int j = wv; j < SEQ / 16; j += 8) {
        const int kv0 = j * 16;
        const v16bf bk0 = load_frag_kmajor(K, HEAD_DIM, kv0, 0,  lane);
        const v16bf bk1 = load_frag_kmajor(K, HEAD_DIM, kv0, 32, lane);
        v8f s = {};
        s = wmma_bf16(aq0, bk0, s);
        s = wmma_bf16(aq1, bk1, s);
        const int c  = lane & 15;
        const int rb = (lane >> 4) * 8;
#pragma unroll
        for (int i = 0; i < 8; ++i)
            sc[rb + i][kv0 + c] = s[i] * INV_SCALE;
    }
    __syncthreads();

    // ---- exact softmax, 2 rows per wave, float4-vectorized columns ----
    for (int rr = 0; rr < 2; ++rr) {
        const int r  = wv * 2 + rr;
        const int c0 = lane * 4;                 // 16B-aligned base
        float mx = -3.0e38f;
        for (int c = c0; c < SEQ; c += 128) {
            const float4 v = *(const float4*)&sc[r][c];
            mx = fmaxf(mx, fmaxf(fmaxf(v.x, v.y), fmaxf(v.z, v.w)));
        }
#pragma unroll
        for (int off = 16; off; off >>= 1) mx = fmaxf(mx, __shfl_xor(mx, off, 32));
        float sum = 0.f;
        for (int c = c0; c < SEQ; c += 128) {
            float4 v = *(const float4*)&sc[r][c];
            v.x = __expf(v.x - mx); v.y = __expf(v.y - mx);
            v.z = __expf(v.z - mx); v.w = __expf(v.w - mx);
            *(float4*)&sc[r][c] = v;
            sum += v.x + v.y + v.z + v.w;
        }
#pragma unroll
        for (int off = 16; off; off >>= 1) sum += __shfl_xor(sum, off, 32);
        const float inv = 1.0f / sum;
        float* arow = attnw + (head * SEQ + (size_t)(q0 + r)) * SEQ;
        for (int c = c0; c < SEQ; c += 128) {
            float4 v = *(const float4*)&sc[r][c];
            v.x *= inv; v.y *= inv; v.z *= inv; v.w *= inv;
            *(float4*)&sc[r][c] = v;
            *(float4*)(arow + c) = v;            // b128 stream of attn_weights
        }
    }
    __syncthreads();

    // ---- context = P @ V ; 8 waves: Dh-tile = wv&3, K-half = wv>>2 ----
    const int n0    = (wv & 3) * 16;
    const int kbeg  = (wv >> 2) * (SEQ / 2);
    const int kend  = kbeg + SEQ / 2;
    const int hi    = (lane >> 4) << 3;
    const int rA    = lane & 15;
    const int cN    = lane & 15;
    const int rb    = (lane >> 4) * 8;
    v8f acc = {};
    for (int k0 = kbeg; k0 < kend; k0 += 32) {
        // A-frag: P rows from LDS (float4 reads), f32 -> bf16 on the fly
        Frag16 fa;
        const float4 x0 = *(const float4*)&sc[rA][k0 + hi];
        const float4 x1 = *(const float4*)&sc[rA][k0 + hi + 4];
        const float4 x2 = *(const float4*)&sc[rA][k0 + 16 + hi];
        const float4 x3 = *(const float4*)&sc[rA][k0 + 16 + hi + 4];
        fa.us[0]  = f32_to_bf16_rne(x0.x); fa.us[1]  = f32_to_bf16_rne(x0.y);
        fa.us[2]  = f32_to_bf16_rne(x0.z); fa.us[3]  = f32_to_bf16_rne(x0.w);
        fa.us[4]  = f32_to_bf16_rne(x1.x); fa.us[5]  = f32_to_bf16_rne(x1.y);
        fa.us[6]  = f32_to_bf16_rne(x1.z); fa.us[7]  = f32_to_bf16_rne(x1.w);
        fa.us[8]  = f32_to_bf16_rne(x2.x); fa.us[9]  = f32_to_bf16_rne(x2.y);
        fa.us[10] = f32_to_bf16_rne(x2.z); fa.us[11] = f32_to_bf16_rne(x2.w);
        fa.us[12] = f32_to_bf16_rne(x3.x); fa.us[13] = f32_to_bf16_rne(x3.y);
        fa.us[14] = f32_to_bf16_rne(x3.z); fa.us[15] = f32_to_bf16_rne(x3.w);
        // B-frag: V^T rows are K-contiguous -> two b128 loads per lane
        const v16bf bv = load_frag_kmajor(V, SEQ, n0, k0, lane);
        acc = wmma_bf16(fa.bf, bv, acc);
    }
    if (wv >= 4) {
#pragma unroll
        for (int i = 0; i < 8; ++i) red[wv - 4][rb + i][cN] = acc[i];
    }
    __syncthreads();
    if (wv < 4) {
#pragma unroll
        for (int i = 0; i < 8; ++i) {
            const float v = acc[i] + red[wv][rb + i][cN];
            const int q = q0 + rb + i;
            ctx[((size_t)b * SEQ + q) * D_MODEL + h * HEAD_DIM + n0 + cN] =
                f32_to_bf16_rne(v);
        }
    }
}

// ---------------------------------------------------------------------------
// Host launcher
// ---------------------------------------------------------------------------
extern "C" void kernel_launch(void* const* d_in, const int* in_sizes, int n_in,
                              void* d_out, int out_size, void* d_ws, size_t ws_size,
                              hipStream_t stream) {
    const float* q_in = (const float*)d_in[0];
    const float* k_in = (const float*)d_in[1];
    const float* v_in = (const float*)d_in[2];
    const float* w_q  = (const float*)d_in[3];
    const float* w_k  = (const float*)d_in[4];
    const float* w_v  = (const float*)d_in[5];
    const float* w_o  = (const float*)d_in[6];
    const float* b_o  = (const float*)d_in[7];

    float* out   = (float*)d_out;                               // [B,S,D]
    float* attnw = (float*)d_out + (size_t)MROWS * D_MODEL;     // [B,H,S,S]

    // Workspace carve-up (bf16 buffers), 256B aligned.
    char* base = (char*)d_ws;
    size_t off = 0;
    auto carve = [&](size_t bytes) -> char* {
        char* p = base + off;
        off += (bytes + 255) & ~(size_t)255;
        return p;
    };
    const size_t actB = (size_t)MROWS * D_MODEL * 2;   // 8 MB
    const size_t wB   = (size_t)D_MODEL * D_MODEL * 2; // 2 MB
    ushort_t* xq  = (ushort_t*)carve(actB);
    ushort_t* xk  = (ushort_t*)carve(actB);
    ushort_t* xv  = (ushort_t*)carve(actB);
    ushort_t* wq2 = (ushort_t*)carve(wB);
    ushort_t* wk2 = (ushort_t*)carve(wB);
    ushort_t* wv2 = (ushort_t*)carve(wB);
    ushort_t* wo2 = (ushort_t*)carve(wB);
    ushort_t* Qh  = (ushort_t*)carve(actB);            // [B,H,S,Dh]
    ushort_t* Kh  = (ushort_t*)carve(actB);            // [B,H,S,Dh]
    ushort_t* Vt  = (ushort_t*)carve(actB);            // [B,H,Dh,S] (transposed)
    ushort_t* ctx = (ushort_t*)carve(actB);            // [B,S,D]
    (void)ws_size; (void)n_in; (void)in_sizes; (void)out_size;

    const int nAct = MROWS * D_MODEL;
    const int nW   = D_MODEL * D_MODEL;
    mha_cvt_bf16<<<2048, 256, 0, stream>>>(q_in, xq, nAct);
    mha_cvt_bf16<<<2048, 256, 0, stream>>>(k_in, xk, nAct);
    mha_cvt_bf16<<<2048, 256, 0, stream>>>(v_in, xv, nAct);
    mha_cvt_bf16<<<1024, 256, 0, stream>>>(w_q, wq2, nW);
    mha_cvt_bf16<<<1024, 256, 0, stream>>>(w_k, wk2, nW);
    mha_cvt_bf16<<<1024, 256, 0, stream>>>(w_v, wv2, nW);
    mha_cvt_bf16<<<1024, 256, 0, stream>>>(w_o, wo2, nW);

    dim3 gemmGrid(MROWS / 16, D_MODEL / 64);           // 256 x 16
    mha_gemm<0><<<gemmGrid, 128, 0, stream>>>(xq, wq2, nullptr, Qh);
    mha_gemm<0><<<gemmGrid, 128, 0, stream>>>(xk, wk2, nullptr, Kh);
    mha_gemm<1><<<gemmGrid, 128, 0, stream>>>(xv, wv2, nullptr, Vt);

    dim3 attnGrid(SEQ / 16, NUM_HEADS, BATCH);         // 128 x 16 x 2
    mha_attn<<<attnGrid, 256, 0, stream>>>(Qh, Kh, Vt, attnw, ctx);

    mha_gemm<2><<<gemmGrid, 128, 0, stream>>>(ctx, wo2, b_o, out);
}
